// RNN_model_43207370998495
// MI455X (gfx1250) — matrix operand order repaired and checked by reading, not compiled, not run on hardware
//
#include <hip/hip_runtime.h>
#include <hip/hip_bf16.h>

typedef __attribute__((ext_vector_type(16))) _Float16 v16h;
typedef __attribute__((ext_vector_type(8)))  float    v8f;

#define HID 15
#define NL  4
#define BT  16     // batch tile per workgroup
#define DIN 64
#define KL0 96     // layer-0 packed K: x(64) + h(15) + pad(17)
#define KLR 32     // layers 1-3 packed K: h_prev(16-padded) + h_rec(15) + pad

// ---- fast activations: use CDNA5 hardware TANH if the builtin exists ----
__device__ __forceinline__ float rcp_sig(float x) {
  return __builtin_amdgcn_rcpf(1.0f + __expf(-x));
}
#if __has_builtin(__builtin_amdgcn_tanhf)
__device__ __forceinline__ float fast_tanh(float x) { return __builtin_amdgcn_tanhf(x); }
__device__ __forceinline__ float fast_sig(float x)  { return 0.5f * fast_tanh(0.5f * x) + 0.5f; }
#else
__device__ __forceinline__ float fast_tanh(float x) { return 2.0f * rcp_sig(2.0f * x) - 1.0f; }
__device__ __forceinline__ float fast_sig(float x)  { return rcp_sig(x); }
#endif

// A fragment (16x32 f16, M x K) in this code's own k order (not AMD's table order):
// lane L: m = L%16, half = L/16; element j: vg=j>>1, lo=j&1,
// K = half*16 + j (contiguous, matches B fragment)
__device__ __forceinline__ v16h load_a_frag(const _Float16* A, int lane, int k0, int kstride) {
  const int m = lane & 15, half = lane >> 4;
  v16h r;
#pragma unroll
  for (int j = 0; j < 16; ++j) {
    const int vg = j >> 1, lo = j & 1;
    const int K = (half << 4) + j; (void)vg; (void)lo;
    r[j] = A[m * kstride + k0 + K];
  }
  return r;
}

// B fragment (32x16 f16, K x N), W stored row-major as [n][K]:
// lane L: n = L%16, half = L/16; element j maps to K = half*16 + j (contiguous)
__device__ __forceinline__ v16h load_b_frag(const _Float16* W, int lane, int k0, int kstride) {
  const int n = lane & 15, half = lane >> 4;
  v16h r;
#pragma unroll
  for (int j = 0; j < 16; ++j) {
    r[j] = W[n * kstride + k0 + (half << 4) + j];
  }
  return r;
}

__global__ __launch_bounds__(128, 1)
void lstm4_fused_wmma(const float* __restrict__ x, const int* __restrict__ lens,
                      const float* __restrict__ Wih0, const float* __restrict__ Wihr,
                      const float* __restrict__ Whh, const float* __restrict__ bih,
                      const float* __restrict__ bhh, float* __restrict__ out,
                      int B, int T) {
  // packed weights: [gate][ncol(16)][K], zero-padded
  __shared__ _Float16 Wc0[4][16][KL0];     // 12 KB
  __shared__ _Float16 WcR[3][4][16][KLR];  // 12 KB
  // persistent A-operand buffers (state lives inside them)
  __shared__ _Float16 A0[BT][KL0];         // [x(0..63) | h0(64..78) | pad]
  __shared__ _Float16 AR[3][BT][KLR];      // [h_{l-1}(0..15 pad) | h_l(16..30) | pad]
  __shared__ float gbuf[4][BT][16];        // cross-wave gate exchange (pre-activation)
  __shared__ float cbuf[NL][BT][16];       // cell state per layer (col 15 = pad, stays 0)
  __shared__ float bsum[NL][4][16];        // fused b_ih+b_hh, gate-padded
  __shared__ int   len_s[BT];

  const int tid  = threadIdx.x;
  const int lane = tid & 31;
  const int wave = tid >> 5;      // wave index == gate index (i,f,g,o)
  const int b0   = blockIdx.x * BT;

  // ---- one-time weight repack into padded f16 LDS layout ----
  for (int i = tid; i < 4 * 16 * KL0; i += 128) {
    const int K = i % KL0; const int n = (i / KL0) & 15; const int g = i / (KL0 * 16);
    float v = 0.f;
    if (n < HID) {
      const int row = g * HID + n;                 // row in [0,60)
      if (K < DIN)            v = Wih0[row * DIN + K];
      else if (K < DIN + HID) v = Whh[row * HID + (K - DIN)];   // layer-0 Whh
    }
    Wc0[g][n][K] = (_Float16)v;
  }
  for (int i = tid; i < 3 * 4 * 16 * KLR; i += 128) {
    const int K = i % KLR; const int n = (i / KLR) & 15;
    const int g = (i / (KLR * 16)) & 3; const int l = i / (KLR * 16 * 4);
    float v = 0.f;
    if (n < HID) {
      const int row = g * HID + n;
      if (K < HID)                        v = Wihr[(l * 60 + row) * HID + K];
      else if (K >= 16 && K < 16 + HID)   v = Whh[((l + 1) * 60 + row) * HID + (K - 16)];
    }
    WcR[l][g][n][K] = (_Float16)v;
  }
  for (int i = tid; i < NL * 4 * 16; i += 128) {
    const int n = i & 15; const int g = (i >> 4) & 3; const int l = i >> 6;
    bsum[l][g][n] = (n < HID) ? (bih[l * 60 + g * HID + n] + bhh[l * 60 + g * HID + n]) : 0.f;
  }
  // zero persistent state (h=0, c=0, padding)
  for (int i = tid; i < BT * KL0;     i += 128) (&A0[0][0])[i]      = (_Float16)0.f;
  for (int i = tid; i < 3 * BT * KLR; i += 128) (&AR[0][0][0])[i]   = (_Float16)0.f;
  for (int i = tid; i < NL * BT * 16; i += 128) (&cbuf[0][0][0])[i] = 0.f;
  if (tid < BT) len_s[tid] = lens[(b0 + tid < B) ? (b0 + tid) : (B - 1)];
  __syncthreads();

  // ---- hoist B fragments + biases into registers (constant over t) ----
  v16h Bf0[3], BfR[3];
#pragma unroll
  for (int c = 0; c < 3; ++c) Bf0[c] = load_b_frag(&Wc0[wave][0][0], lane, c * 32, KL0);
#pragma unroll
  for (int l = 0; l < 3; ++l) BfR[l] = load_b_frag(&WcR[l][wave][0][0], lane, 0, KLR);
  float bl[NL];
#pragma unroll
  for (int l = 0; l < NL; ++l) bl[l] = bsum[l][wave][lane & 15];

  // x staging assignment: thread -> (row, 8-float segment)
  const int  xrow  = tid >> 3;         // 0..15
  const int  xcol  = (tid & 7) * 8;    // 0..56
  const long rbase = (long)(b0 + xrow) * (long)T * DIN + xcol;

  // elementwise slot assignment: 2 slots/thread, no divergence
  const int erow0 = tid >> 4, eu = tid & 15;  // slot tid:  (erow0, eu)
                                              // slot tid+128: (erow0+8, eu)

  for (int t = 0; t < T; ++t) {
    // stage x_t tile fp32 -> f16 into A0[:, 0..63]; prefetch t+1
    const float* src = x + rbase + (long)t * DIN;
#pragma unroll
    for (int j = 0; j < 8; ++j) A0[xrow][xcol + j] = (_Float16)src[j];
    if (t + 1 < T) __builtin_prefetch((const void*)(src + DIN), 0, 1);
    __syncthreads();

#pragma unroll
    for (int l = 0; l < NL; ++l) {
      // gates for this wave's gate index, bias pre-loaded into accumulator
      v8f acc;
#pragma unroll
      for (int r = 0; r < 8; ++r) acc[r] = bl[l];
      if (l == 0) {
#pragma unroll
        for (int c = 0; c < 3; ++c) {
          v16h a = load_a_frag(&A0[0][0], lane, c * 32, KL0);
          acc = __builtin_amdgcn_wmma_f32_16x16x32_f16(
                    false, a, false, Bf0[c], (short)0, acc, false, false);
        }
      } else {
        v16h a = load_a_frag(&AR[l - 1][0][0], lane, 0, KLR);
        acc = __builtin_amdgcn_wmma_f32_16x16x32_f16(
                  false, a, false, BfR[l - 1], (short)0, acc, false, false);
      }
      // scatter this wave's 16x16 gate tile (C/D layout: row = r + 8*half, col = lane%16)
      {
        const int col = lane & 15, half = lane >> 4;
#pragma unroll
        for (int r = 0; r < 8; ++r) gbuf[wave][r + half * 8][col] = acc[r];
      }
      __syncthreads();

      // elementwise LSTM cell: 256 padded slots, 2 per thread, fully uniform.
      // Pad slots (u==15) provably stay zero (zero weights + zero bias).
#pragma unroll
      for (int it = 0; it < 2; ++it) {
        const int row = erow0 + it * 8, u = eu;
        const float gi = fast_sig(gbuf[0][row][u]);
        const float gf = fast_sig(gbuf[1][row][u]);
        const float gg = fast_tanh(gbuf[2][row][u]);
        const float go = fast_sig(gbuf[3][row][u]);
        const float cv = gf * cbuf[l][row][u] + gi * gg;
        cbuf[l][row][u] = cv;
        const float h = go * fast_tanh(cv);
        const _Float16 h16 = (_Float16)h;
        if (l == 0) {
          A0[row][DIN + u] = h16;        // recurrent slot (u==15 -> zero pad col 79)
          AR[0][row][u]    = h16;        // feeds layer 1 (col 15 = zero-weight pad)
        } else {
          AR[l - 1][row][16 + u] = h16;  // recurrent slot (col 31 = pad)
          if (l < NL - 1) {
            AR[l][row][u] = h16;         // feeds layer l+1
          } else if (u < HID) {
            const int b = b0 + row;      // final layer: gather at t == len-1
            if (b < B && t == len_s[row] - 1) out[b * HID + u] = h;
          }
        }
      }
      __syncthreads();
    }
  }
}

extern "C" void kernel_launch(void* const* d_in, const int* in_sizes, int n_in,
                              void* d_out, int out_size, void* d_ws, size_t ws_size,
                              hipStream_t stream) {
  const float* emb  = (const float*)d_in[0];  // [B, T, 64]
  const int*   lens = (const int*)  d_in[1];  // [B]
  const float* Wih0 = (const float*)d_in[2];  // [60, 64]
  const float* Wihr = (const float*)d_in[3];  // [3, 60, 15]
  const float* Whh  = (const float*)d_in[4];  // [4, 60, 15]
  const float* bih  = (const float*)d_in[5];  // [4, 60]
  const float* bhh  = (const float*)d_in[6];  // [4, 60]
  float* out = (float*)d_out;                 // [B, 15]

  const int B = in_sizes[1];
  const int T = in_sizes[0] / (B * DIN);
  const int nblocks = (B + BT - 1) / BT;

  lstm4_fused_wmma<<<nblocks, 128, 0, stream>>>(emb, lens, Wih0, Wihr, Whh,
                                                bih, bhh, out, B, T);
}
